// GLAttLayer_83820581749505
// MI455X (gfx1250) — matrix operand order repaired and checked by reading, not compiled, not run on hardware
//
#include <hip/hip_runtime.h>
#include <hip/hip_bf16.h>
#include <math.h>

// ---------------------------------------------------------------------------
// MI455X (gfx1250): all GEMMs via v_wmma_f32_16x16x32_f16, GEMM tiles staged
// by the Tensor Data Mover (tensor_load_to_lds + s_wait_tensorcnt), fused
// window attention with linearized LDS bias lookup and vectorized fragments.
// ---------------------------------------------------------------------------

typedef _Float16 h16;
typedef __attribute__((ext_vector_type(16))) _Float16 v16h;
typedef __attribute__((ext_vector_type(8)))  _Float16 v8h;
typedef __attribute__((ext_vector_type(8)))  float    v8f;
typedef unsigned long long u64x;

union FragA  { v16h v; h16   e[16]; u64x q[4]; };
union FragH8 { v8h  v; h16   e[8];  u64x q[2]; };
union FragC  { v8f  v; float e[8];  };
union F4     { float4 v; float e[4]; };

#define NTOKC 175616   // 56^3
#define CD    96

// ---- TDM availability (device pass only; fallback = cooperative staging) ---
#if defined(__HIP_DEVICE_COMPILE__) && defined(__gfx1250__) && defined(__has_builtin)
#if __has_builtin(__builtin_amdgcn_tensor_load_to_lds)
#define USE_TDM 1
#endif
#endif
#ifndef USE_TDM
#define USE_TDM 0
#endif

#if USE_TDM
typedef unsigned int v4u __attribute__((ext_vector_type(4)));
typedef int          v4i __attribute__((ext_vector_type(4)));
typedef int          v8i __attribute__((ext_vector_type(8)));

// 2D tile load via Tensor Data Mover. dims/strides in elements (data_size=2B).
__device__ __forceinline__ void tdm_load2d(unsigned lds_addr, const void* gp,
                                           unsigned dim0, unsigned dim1,
                                           unsigned tile0, unsigned tile1,
                                           unsigned long long stride,
                                           unsigned padcfg) {
  unsigned long long ga = (unsigned long long)(uintptr_t)gp;
  v4u g0;
  g0[0] = 1u;                                   // count=1, user mode
  g0[1] = lds_addr;                             // LDS byte address
  g0[2] = (unsigned)ga;                         // global_addr[31:0]
  g0[3] = (unsigned)((ga >> 32) & 0x01FFFFFFu) | (2u << 30);  // addr[56:32]|type=2
  v8i g1;
  g1[0] = (int)(0x00010000u | padcfg);          // data_size=1 (2B) + padding
  g1[1] = (int)((dim0 & 0xFFFFu) << 16);        // [63:48] tensor_dim0 lo
  g1[2] = (int)(((dim0 >> 16) & 0xFFFFu) | ((dim1 & 0xFFFFu) << 16));
  g1[3] = (int)(((dim1 >> 16) & 0xFFFFu) | (tile0 << 16));
  g1[4] = (int)(tile1 & 0xFFFFu);               // tile_dim1 (tile_dim2=0)
  g1[5] = (int)(unsigned)stride;                // tensor_dim0_stride lo
  g1[6] = (int)((unsigned)(stride >> 32) & 0xFFFFu);
  g1[7] = 0;
  v4i z4 = {0, 0, 0, 0};
#if __clang_major__ >= 23
  v8i z8 = {0, 0, 0, 0, 0, 0, 0, 0};
  __builtin_amdgcn_tensor_load_to_lds(g0, g1, z4, z4, z8, 0);
#else
  __builtin_amdgcn_tensor_load_to_lds(g0, g1, z4, z4, 0);
#endif
}
__device__ __forceinline__ void tdm_wait() {
#if __has_builtin(__builtin_amdgcn_s_wait_tensorcnt)
  __builtin_amdgcn_s_wait_tensorcnt(0);
#else
  asm volatile("s_wait_tensorcnt 0" ::: "memory");
#endif
}
#define PAD_R36 ((1u << 20) | (3u << 22) | (1u << 25))  // 16 dw + 2 dw pad -> stride 36 h
#define PAD_R66 ((1u << 20) | (4u << 22) | (0u << 25))  // 32 dw + 1 dw pad -> stride 66 h
#endif  // USE_TDM

__device__ __forceinline__ v8f wmma16(v16h a, v16h b, v8f c) {
  return __builtin_amdgcn_wmma_f32_16x16x32_f16(false, a, false, b, (short)0, c, false, false);
}

__device__ __forceinline__ int akidx(int j, int p, int hf) {
  return ((j >> 2) << 4) + ((j & 3) << 1) + (hf << 3) + p;
}
__device__ __forceinline__ int bkidx(int j, int p, int hf) {
  return (hf << 4) + (j << 1) + p;
}

__device__ __forceinline__ int g2nat(int g) {
  int win = g / 343, idx = g % 343;
  int ws_ = win >> 6, wh = (win >> 3) & 7, ww = win & 7;
  int is = idx / 49, ih = (idx / 7) % 7, iw = idx % 7;
  int s = ws_ * 7 + is, h = wh * 7 + ih, w = ww * 7 + iw;
  return (s * 56 + h) * 56 + w;
}
__device__ __forceinline__ int nat2g(int n) {
  int s = n / 3136, h = (n / 56) % 56, w = n % 56;
  int win = (s / 7) * 64 + (h / 7) * 8 + (w / 7);
  int idx = (s % 7) * 49 + (h % 7) * 7 + (w % 7);
  return win * 343 + idx;
}

__device__ __forceinline__ float gelu_f(float v) {
  return 0.5f * v * (1.0f + erff(v * 0.70710678118654752f));
}
__device__ __forceinline__ float wred_max(float v) {
  for (int m = 16; m > 0; m >>= 1) v = fmaxf(v, __shfl_xor(v, m, 32));
  return v;
}
__device__ __forceinline__ float wred_sum(float v) {
  for (int m = 16; m > 0; m >>= 1) v += __shfl_xor(v, m, 32);
  return v;
}
__device__ __forceinline__ int clampi(int v, int lo, int hi) {
  return v < lo ? lo : (v > hi ? hi : v);
}

// ---------------------------------------------------------------------------
// Generic tiled WMMA GEMM, 128(M)x64(N) tile, K-step 32, 8 waves.
// TDM double-buffered staging when available, cooperative loads otherwise.
// ---------------------------------------------------------------------------
template <int OUT_F16, int TRANS_B, int ACT, int HAS_BIAS, int RES_MODE, int STORE_MODE>
__global__ __launch_bounds__(256) void gemm_wmma(
    const h16* __restrict__ A, long lda, long sA,
    const h16* __restrict__ B, long ldb, long sB,
    void* __restrict__ Cp, long ldc, long sC,
    const float* __restrict__ bias,
    const float* __restrict__ Res, long ldr,
    const float* __restrict__ bng, const float* __restrict__ bnb,
    float alpha, int M, int N, int K) {
  __shared__ __align__(16) h16 As[2][128 * 36];
  __shared__ __align__(16) h16 Bs[2][2304];  // max(64*36, 32*66)
  const int tid = threadIdx.x;
  const int wave = tid >> 5, lane = tid & 31;
  const int n0 = blockIdx.x * 64, m0 = blockIdx.y * 128;
  const int z = blockIdx.z;
  A += (long)z * sA;
  B += (long)z * sB;

  FragC acc[4];
#pragma unroll
  for (int i = 0; i < 4; i++)
#pragma unroll
    for (int v = 0; v < 8; v++) acc[i].e[v] = 0.f;

  const int r = lane & 15, hf = lane >> 4;
  const int nk = (K + 31) / 32;
  int buf = 0;

#if USE_TDM
  if (wave == 0) {
    tdm_load2d((unsigned)(uintptr_t)&As[0][0], A + (long)m0 * lda,
               (unsigned)K, (unsigned)(M - m0), 32u, 128u,
               (unsigned long long)lda, PAD_R36);
    if (TRANS_B)
      tdm_load2d((unsigned)(uintptr_t)&Bs[0][0], B + (long)n0 * ldb,
                 (unsigned)K, (unsigned)(N - n0), 32u, 64u,
                 (unsigned long long)ldb, PAD_R36);
    else
      tdm_load2d((unsigned)(uintptr_t)&Bs[0][0], B + n0,
                 (unsigned)(N - n0), (unsigned)K, 64u, 32u,
                 (unsigned long long)ldb, PAD_R66);
  }
#endif

  for (int s = 0; s < nk; ++s) {
    const int k0 = s * 32;
#if USE_TDM
    if (wave == 0) tdm_wait();
    __syncthreads();
    if (wave == 0 && s + 1 < nk) {
      const int kn = k0 + 32;
      tdm_load2d((unsigned)(uintptr_t)&As[buf ^ 1][0], A + (long)m0 * lda + kn,
                 (unsigned)(K - kn), (unsigned)(M - m0), 32u, 128u,
                 (unsigned long long)lda, PAD_R36);
      if (TRANS_B)
        tdm_load2d((unsigned)(uintptr_t)&Bs[buf ^ 1][0], B + (long)n0 * ldb + kn,
                   (unsigned)(K - kn), (unsigned)(N - n0), 32u, 64u,
                   (unsigned long long)ldb, PAD_R36);
      else
        tdm_load2d((unsigned)(uintptr_t)&Bs[buf ^ 1][0], B + (long)kn * ldb + n0,
                   (unsigned)(N - n0), (unsigned)(K - kn), 64u, 32u,
                   (unsigned long long)ldb, PAD_R66);
    }
#else
    for (int i = tid; i < 4096; i += 256) {
      int rr = i >> 5, kk = i & 31;
      int gr = m0 + rr, gk = k0 + kk;
      As[buf][rr * 36 + kk] = (gr < M && gk < K) ? A[(long)gr * lda + gk] : (h16)0;
    }
    if (TRANS_B) {
      for (int i = tid; i < 2048; i += 256) {
        int nn = i >> 5, kk = i & 31;
        int gn = n0 + nn, gk = k0 + kk;
        Bs[buf][nn * 36 + kk] = (gn < N && gk < K) ? B[(long)gn * ldb + gk] : (h16)0;
      }
    } else {
      for (int i = tid; i < 2048; i += 256) {
        int kk = i >> 6, nn = i & 63;
        int gk = k0 + kk, gn = n0 + nn;
        Bs[buf][kk * 66 + nn] = (gk < K && gn < N) ? B[(long)gk * ldb + gn] : (h16)0;
      }
    }
    __syncthreads();
#endif

    FragA af;
    {
      int row = wave * 16 + r;
      const u64x* c0 = (const u64x*)&As[buf][row * 36 + 16 * hf];
      af.q[0] = c0[0];
      af.q[1] = c0[1];
      af.q[2] = c0[2];
      af.q[3] = c0[3];
    }
#pragma unroll
    for (int ns = 0; ns < 4; ns++) {
      FragA bf;
      if (TRANS_B) {
        int nn = ns * 16 + r;
        const u64x* cb = (const u64x*)&Bs[buf][nn * 36 + 16 * hf];
        bf.q[0] = cb[0];
        bf.q[1] = cb[1];
        bf.q[2] = cb[2];
        bf.q[3] = cb[3];
      } else {
#pragma unroll
        for (int j = 0; j < 8; j++)
#pragma unroll
          for (int p = 0; p < 2; p++)
            bf.e[2 * j + p] = Bs[buf][bkidx(j, p, hf) * 66 + ns * 16 + r];
      }
      acc[ns].v = wmma16(af.v, bf.v, acc[ns].v);
    }
    __syncthreads();
    buf ^= 1;
  }

#pragma unroll
  for (int ns = 0; ns < 4; ns++) {
    int gcol = n0 + ns * 16 + r;
    if (gcol >= N) continue;
#pragma unroll
    for (int v = 0; v < 8; v++) {
      int grow = m0 + wave * 16 + v + (hf << 3);
      if (grow >= M) continue;
      float val = alpha * acc[ns].e[v];
      if (HAS_BIAS) val += bias[gcol];
      if (ACT == 1) val = gelu_f(val);
      if (RES_MODE == 1) val += Res[(long)grow * ldr + gcol];
      if (RES_MODE == 2) val += Res[(long)gcol * NTOKC + g2nat(grow)];
      if (STORE_MODE == 0) {
        if (OUT_F16)
          ((h16*)Cp)[(long)z * sC + (long)grow * ldc + gcol] = (h16)val;
        else
          ((float*)Cp)[(long)z * sC + (long)grow * ldc + gcol] = val;
      } else {
        float y = val * (bng[gcol] * 0.9999950000374997f) + bnb[gcol];  // BN eval
        y = (y >= 0.f) ? y : 0.01f * y;                                  // LeakyReLU
        ((float*)Cp)[(long)gcol * NTOKC + grow] = y;
      }
    }
  }
}

// ---------------------------------------------------------------------------
__global__ __launch_bounds__(256) void k_local_ln(const float* __restrict__ x,
                                                  const float* __restrict__ w,
                                                  const float* __restrict__ b,
                                                  h16* __restrict__ out) {
  int n = blockIdx.x * 256 + threadIdx.x;
  float s = 0.f, ss = 0.f;
  for (int c = 0; c < CD; c++) {
    float v = x[(long)c * NTOKC + n];
    s += v;
    ss += v * v;
  }
  float mu = s * (1.f / CD);
  float var = ss * (1.f / CD) - mu * mu;
  float rs = rsqrtf(var + 1e-5f);
  long g = nat2g(n);
  for (int c = 0; c < CD; c++)
    out[g * CD + c] = (h16)((x[(long)c * NTOKC + n] - mu) * rs * w[c] + b[c]);
}

__global__ __launch_bounds__(256) void k_ln_rows(const float* __restrict__ in,
                                                 const float* __restrict__ w,
                                                 const float* __restrict__ b,
                                                 h16* __restrict__ out, int M) {
  int wave = threadIdx.x >> 5, lane = threadIdx.x & 31;
  int row = blockIdx.x * 8 + wave;
  if (row >= M) return;
  const float* p = in + (long)row * CD;
  float s = 0.f, ss = 0.f;
  for (int c = lane; c < CD; c += 32) {
    float v = p[c];
    s += v;
    ss += v * v;
  }
  s = wred_sum(s);
  ss = wred_sum(ss);
  float mu = s * (1.f / CD);
  float var = ss * (1.f / CD) - mu * mu;
  float rs = rsqrtf(var + 1e-5f);
  for (int c = lane; c < CD; c += 32)
    out[(long)row * CD + c] = (h16)((p[c] - mu) * rs * w[c] + b[c]);
}

// ---------------------------------------------------------------------------
// Fused window attention, one block per (window, head).
// Relative-position bias index is LINEAR in the packed digit value:
//   bidx = Bq - Bk + 1098,  B(t) = (t/49)*169 + ((t/7)%7)*13 + (t%7)
// so the inner loop is branch-free: one Bk read per key subtile + clamped
// LDS bias gathers; query-side Bq values are hoisted per row-tile.
// ---------------------------------------------------------------------------
__global__ __launch_bounds__(256) void k_win_attn(const h16* __restrict__ qkv,
                                                  const float* __restrict__ rb,
                                                  h16* __restrict__ ao) {
  __shared__ __align__(16) float sc[16][352];
  __shared__ __align__(32) h16 sq[512];       // A-fragment order: lane*16+e
  __shared__ __align__(32) h16 kv[8][1024];   // per-wave B-fragment staging
  __shared__ float outs[16 * 32];
  __shared__ float rbl[2197];                 // per-head bias table
  __shared__ int bq[352];                     // packed digit base per token
  const int tid = threadIdx.x, wave = tid >> 5, lane = tid & 31;
  const int win = blockIdx.x / 3, head = blockIdx.x % 3;
  const h16* qb = qkv + (long)win * 343 * 288 + head * 32;
  const h16* kb = qb + 96;
  const h16* vb = qb + 192;
  const float scale = 0.17677669529663687f;  // 32^-0.5
  const int r = lane & 15, hf = lane >> 4;

  for (int i = tid; i < 2197; i += 256) rbl[i] = rb[i * 3 + head];
  for (int i = tid; i < 352; i += 256) {
    int s = i / 49, hh = (i / 7) % 7, ww = i % 7;
    bq[i] = s * 169 + hh * 13 + ww;
  }
  const int e_l = 2 * ((lane >> 1) & 7) + (lane & 1);  // B-frag elem for K=lane

  for (int rt = 0; rt < 22; rt++) {
    int m0 = rt * 16;
    __syncthreads();
    // stage q tile directly in A-fragment order (8-half chunks)
    if (tid < 64) {
      int lt = tid & 31, chunk = tid >> 5;
      int row = lt & 15, lhf = lt >> 4;
      int gr = m0 + row;
      int d0 = lhf * 16 + 8 * chunk;
      FragH8 vq;
      if (gr < 343)
        vq.v = *(const v8h*)(qb + (long)gr * 288 + d0);
      else
        for (int t = 0; t < 8; t++) vq.e[t] = (h16)0;
      *(v8h*)&sq[lt * 16 + chunk * 8] = vq.v;
    }
    __syncthreads();

    FragA af;
    af.v = *(const v16h*)&sq[lane * 16];

    // hoist the 8 query-side digit bases (+ validity) for this row tile
    int qB[8];
    bool qok[8];
#pragma unroll
    for (int v = 0; v < 8; v++) {
      int qn = m0 + v + (hf << 3);
      qok[v] = qn < 343;
      qB[v] = bq[qok[v] ? qn : 342];
    }

    // scores: wave-strided key subtiles
    for (int nt = wave; nt < 22; nt += 8) {
      int n0 = nt * 16;
      int gtok = n0 + r;
      FragA kf;
      if (gtok < 343)
        kf.v = *(const v16h*)(kb + (long)gtok * 288 + hf * 16);
      else
        for (int t = 0; t < 16; t++) kf.e[t] = (h16)0;
      *(v16h*)&kv[wave][lane * 16] = kf.v;
      FragA bf;
      bf.v = *(const v16h*)&kv[wave][lane * 16];
      FragC acc;
#pragma unroll
      for (int v = 0; v < 8; v++) acc.e[v] = 0.f;
      acc.v = wmma16(af.v, bf.v, acc.v);
      int m = n0 + r;
      bool mok = m < 343;
      int Bk = bq[m];
#pragma unroll
      for (int v = 0; v < 8; v++) {
        int bidx = clampi(qB[v] - Bk + 1098, 0, 2196);
        float bias = rbl[bidx];                       // unconditional LDS gather
        float sv = (mok && qok[v]) ? acc.e[v] * scale + bias : 0.f;
        sc[v + (hf << 3)][m] = sv;
      }
    }
    __syncthreads();

    // softmax over 343 keys (wave handles 2 rows)
    for (int rr = 0; rr < 2; rr++) {
      int rw = wave * 2 + rr;
      float mx = -1e30f;
      for (int j = lane; j < 343; j += 32) mx = fmaxf(mx, sc[rw][j]);
      mx = wred_max(mx);
      float sm = 0.f;
      for (int j = lane; j < 343; j += 32) {
        float e = __expf(sc[rw][j] - mx);
        sc[rw][j] = e;
        sm += e;
      }
      sm = wred_sum(sm);
      float inv = 1.f / sm;
      for (int j = lane; j < 343; j += 32) sc[rw][j] *= inv;
    }
    for (int i = tid; i < 512; i += 256) outs[i] = 0.f;
    __syncthreads();

    // P @ V : wave-strided K-steps, cross-wave reduce via ds_add_f32
    for (int ks = wave; ks < 11; ks += 8) {
      int t0 = ks * 32;
      int gv = t0 + lane;
#pragma unroll
      for (int d = 0; d < 32; d++) {
        h16 val = (gv < 343) ? vb[(long)gv * 288 + d] : (h16)0;
        int sub = d >> 4;
        int dl = (d & 15) + ((lane >> 4) << 4);
        kv[wave][sub * 512 + dl * 16 + e_l] = val;
      }
      // P fragment: two contiguous 8-float runs -> 4x ds_load_b128 + cvt
      FragA af2;
      {
        const float4* p1 = (const float4*)&sc[r][t0 + 16 * hf];
        const float4* p2 = (const float4*)&sc[r][t0 + 16 * hf + 8];
        F4 a0, a1, a2, a3;
        a0.v = p1[0];
        a1.v = p1[1];
        a2.v = p2[0];
        a3.v = p2[1];
#pragma unroll
        for (int t = 0; t < 4; t++) {
          af2.e[t] = (h16)a0.e[t];
          af2.e[4 + t] = (h16)a1.e[t];
          af2.e[8 + t] = (h16)a2.e[t];
          af2.e[12 + t] = (h16)a3.e[t];
        }
      }
#pragma unroll
      for (int sub = 0; sub < 2; sub++) {
        FragA bf;
        bf.v = *(const v16h*)&kv[wave][sub * 512 + lane * 16];
        FragC acc;
#pragma unroll
        for (int v = 0; v < 8; v++) acc.e[v] = 0.f;
        acc.v = wmma16(af2.v, bf.v, acc.v);
#pragma unroll
        for (int v = 0; v < 8; v++)
          atomicAdd(&outs[(v + (hf << 3)) * 32 + sub * 16 + r], acc.e[v]);
      }
    }
    __syncthreads();

    for (int i = tid; i < 512; i += 256) {
      int rr = i >> 5, d = i & 31;
      int gr = m0 + rr;
      if (gr < 343)
        ao[((long)win * 343 + gr) * CD + head * 32 + d] = (h16)outs[rr * 32 + d];
    }
  }
}

// ---------------------------------------------------------------------------
__global__ __launch_bounds__(256) void k_pool(const float* __restrict__ x,
                                              float* __restrict__ tg) {
  int idx = blockIdx.x * 256 + threadIdx.x;  // 96*1728 exact
  int c = idx / 1728, tok = idx % 1728;
  int p = tok / 144, q = (tok / 12) % 12, rr = tok % 12;
  int s0 = (p * 56) / 12, s1 = ((p + 1) * 56 + 11) / 12;
  int h0 = (q * 56) / 12, h1 = ((q + 1) * 56 + 11) / 12;
  int w0 = (rr * 56) / 12, w1 = ((rr + 1) * 56 + 11) / 12;
  const float* xc = x + (long)c * NTOKC;
  float sum = 0.f;
  for (int s = s0; s < s1; s++)
    for (int h = h0; h < h1; h++)
      for (int w = w0; w < w1; w++) sum += xc[(s * 56 + h) * 56 + w];
  float inv = 1.f / (float)((s1 - s0) * (h1 - h0) * (w1 - w0));
  tg[(long)tok * CD + c] = sum * inv;
}

__global__ __launch_bounds__(256) void k_softmax(const float* __restrict__ P,
                                                 h16* __restrict__ O) {
  long row = blockIdx.x;
  const float* p = P + row * 1728;
  __shared__ float wb[8];
  __shared__ float bc;
  int tid = threadIdx.x, wave = tid >> 5, lane = tid & 31;
  float mx = -1e30f;
  for (int j = tid; j < 1728; j += 256) mx = fmaxf(mx, p[j]);
  mx = wred_max(mx);
  if (lane == 0) wb[wave] = mx;
  __syncthreads();
  if (tid == 0) {
    float m = wb[0];
    for (int i = 1; i < 8; i++) m = fmaxf(m, wb[i]);
    bc = m;
  }
  __syncthreads();
  mx = bc;
  float sm = 0.f;
  for (int j = tid; j < 1728; j += 256) sm += __expf(p[j] - mx);
  sm = wred_sum(sm);
  if (lane == 0) wb[wave] = sm;
  __syncthreads();
  if (tid == 0) {
    float t = 0.f;
    for (int i = 0; i < 8; i++) t += wb[i];
    bc = t;
  }
  __syncthreads();
  float inv = 1.f / bc;
  for (int j = tid; j < 1728; j += 256) O[row * 1728 + j] = (h16)(__expf(p[j] - mx) * inv);
}

// ---------------------------------------------------------------------------
__device__ __forceinline__ void triaxis(int o, int& i0, int& i1, float& f) {
  float pos = (o + 0.5f) * (12.f / 56.f) - 0.5f;
  float fl = floorf(pos);
  f = pos - fl;
  int i = (int)fl;
  i0 = clampi(i, 0, 11);
  i1 = clampi(i + 1, 0, 11);
}

__global__ __launch_bounds__(256) void k_cat(const float* __restrict__ t2,
                                             const float* __restrict__ tg3,
                                             h16* __restrict__ cat) {
  int n = blockIdx.x * 256 + threadIdx.x;
  int s = n / 3136, h = (n / 56) % 56, w = n % 56;
  long g = nat2g(n);
  h16* row = cat + (long)n * 192;
  for (int c = 0; c < CD; c++) row[c] = (h16)t2[g * CD + c];
  int s0, s1, h0, h1, w0, w1;
  float fs, fh, fw;
  triaxis(s, s0, s1, fs);
  triaxis(h, h0, h1, fh);
  triaxis(w, w0, w1, fw);
  long base[8];
  float wt[8];
  int pi[2] = {s0, s1}, qi[2] = {h0, h1}, ri[2] = {w0, w1};
  float wp[2] = {1.f - fs, fs}, wq[2] = {1.f - fh, fh}, wr[2] = {1.f - fw, fw};
  int t = 0;
  for (int a = 0; a < 2; a++)
    for (int bI = 0; bI < 2; bI++)
      for (int cI = 0; cI < 2; cI++) {
        base[t] = (long)((pi[a] * 12 + qi[bI]) * 12 + ri[cI]) * CD;
        wt[t] = wp[a] * wq[bI] * wr[cI];
        t++;
      }
  for (int c = 0; c < CD; c++) {
    float v = 0.f;
    for (int k = 0; k < 8; k++) v += wt[k] * tg3[base[k] + c];
    row[96 + c] = (h16)v;
  }
}

__global__ void k_f32_to_f16(const float* __restrict__ a, h16* __restrict__ b, int n) {
  int i = blockIdx.x * 256 + threadIdx.x;
  if (i < n) b[i] = (h16)a[i];
}

// ---------------------------------------------------------------------------
extern "C" void kernel_launch(void* const* d_in, const int* in_sizes, int n_in,
                              void* d_out, int out_size, void* d_ws, size_t ws_size,
                              hipStream_t stream) {
  const float* x = (const float*)d_in[0];
  const float* ln1_w = (const float*)d_in[1];
  const float* ln1_b = (const float*)d_in[2];
  const float* qkv_w = (const float*)d_in[3];
  const float* qkv_b = (const float*)d_in[4];
  const float* rel_bias = (const float*)d_in[5];
  const float* proj_w = (const float*)d_in[6];
  const float* proj_b = (const float*)d_in[7];
  const float* ln2_w = (const float*)d_in[8];
  const float* ln2_b = (const float*)d_in[9];
  const float* mlp1_w = (const float*)d_in[10];
  const float* mlp1_b = (const float*)d_in[11];
  const float* mlp2_w = (const float*)d_in[12];
  const float* mlp2_b = (const float*)d_in[13];
  const float* g_ln1_w = (const float*)d_in[14];
  const float* g_ln1_b = (const float*)d_in[15];
  const float* q_w = (const float*)d_in[16];
  const float* q_b = (const float*)d_in[17];
  const float* k_w = (const float*)d_in[18];
  const float* k_b = (const float*)d_in[19];
  const float* v_w = (const float*)d_in[20];
  const float* v_b = (const float*)d_in[21];
  const float* o_w = (const float*)d_in[22];
  const float* o_b = (const float*)d_in[23];
  const float* g_ln2_w = (const float*)d_in[24];
  const float* g_ln2_b = (const float*)d_in[25];
  const float* g_fc1_w = (const float*)d_in[26];
  const float* g_fc1_b = (const float*)d_in[27];
  const float* g_fc2_w = (const float*)d_in[28];
  const float* g_fc2_b = (const float*)d_in[29];
  const float* conv_w = (const float*)d_in[30];
  const float* conv_b = (const float*)d_in[31];
  const float* bn_g = (const float*)d_in[32];
  const float* bn_b = (const float*)d_in[33];
  (void)in_sizes; (void)n_in; (void)out_size; (void)ws_size;

  char* ws = (char*)d_ws;
  size_t off = 0;
  auto alloc = [&](size_t bytes) {
    size_t o = off;
    off = (off + bytes + 255) & ~(size_t)255;
    return (void*)(ws + o);
  };
  const long NT = NTOKC, GP = 1728;
  void* r1 = alloc((size_t)NT * 384 * 2);  // qkv_g -> hid -> cat
  void* r2 = alloc((size_t)NT * 96 * 4);   // t -> P
  void* r3 = alloc((size_t)NT * 96 * 4);   // t2
  void* r4 = alloc((size_t)NT * 96 * 2);   // tln -> attn -> u -> P16
  float* tg   = (float*)alloc((size_t)GP * 96 * 4);
  h16*   ag16 = (h16*)  alloc((size_t)GP * 96 * 2);
  h16*   qg   = (h16*)  alloc((size_t)GP * 96 * 2);
  h16*   kg   = (h16*)  alloc((size_t)GP * 96 * 2);
  h16*   vg   = (h16*)  alloc((size_t)GP * 96 * 2);
  h16*   ago  = (h16*)  alloc((size_t)GP * 96 * 2);
  float* tg2  = (float*)alloc((size_t)GP * 96 * 4);
  h16*   a2   = (h16*)  alloc((size_t)GP * 96 * 2);
  h16*   h2   = (h16*)  alloc((size_t)GP * 96 * 2);
  float* tg3  = (float*)alloc((size_t)GP * 96 * 4);
  h16* w_qkv = (h16*)alloc(288 * 96 * 2);
  h16* w_prj = (h16*)alloc(96 * 96 * 2);
  h16* w_m1  = (h16*)alloc(384 * 96 * 2);
  h16* w_m2  = (h16*)alloc(96 * 384 * 2);
  h16* w_q   = (h16*)alloc(96 * 96 * 2);
  h16* w_k   = (h16*)alloc(96 * 96 * 2);
  h16* w_v   = (h16*)alloc(96 * 96 * 2);
  h16* w_o   = (h16*)alloc(96 * 96 * 2);
  h16* w_f1  = (h16*)alloc(96 * 96 * 2);
  h16* w_f2  = (h16*)alloc(96 * 96 * 2);
  h16* w_cv  = (h16*)alloc(96 * 192 * 2);

  auto cvt = [&](const float* s, h16* d, int n) {
    k_f32_to_f16<<<(n + 255) / 256, 256, 0, stream>>>(s, d, n);
  };
  cvt(qkv_w, w_qkv, 288 * 96);
  cvt(proj_w, w_prj, 96 * 96);
  cvt(mlp1_w, w_m1, 384 * 96);
  cvt(mlp2_w, w_m2, 96 * 384);
  cvt(q_w, w_q, 96 * 96);
  cvt(k_w, w_k, 96 * 96);
  cvt(v_w, w_v, 96 * 96);
  cvt(o_w, w_o, 96 * 96);
  cvt(g_fc1_w, w_f1, 96 * 96);
  cvt(g_fc2_w, w_f2, 96 * 96);
  cvt(conv_w, w_cv, 96 * 192);

  auto grid = [](int M, int N, int bat) {
    return dim3((unsigned)((N + 63) / 64), (unsigned)((M + 127) / 128), (unsigned)bat);
  };

  // ---------------- local branch ----------------
  k_local_ln<<<NT / 256, 256, 0, stream>>>(x, ln1_w, ln1_b, (h16*)r4);
  gemm_wmma<1, 1, 0, 1, 0, 0><<<grid(NT, 288, 1), 256, 0, stream>>>(
      (const h16*)r4, 96, 0, w_qkv, 96, 0, r1, 288, 0, qkv_b, nullptr, 0,
      nullptr, nullptr, 1.f, NT, 288, 96);
  k_win_attn<<<512 * 3, 256, 0, stream>>>((const h16*)r1, rel_bias, (h16*)r4);
  gemm_wmma<0, 1, 0, 1, 2, 0><<<grid(NT, 96, 1), 256, 0, stream>>>(
      (const h16*)r4, 96, 0, w_prj, 96, 0, r2, 96, 0, proj_b, x, 0,
      nullptr, nullptr, 1.f, NT, 96, 96);
  k_ln_rows<<<(NT / 8), 256, 0, stream>>>((const float*)r2, ln2_w, ln2_b, (h16*)r4, NT);
  gemm_wmma<1, 1, 1, 1, 0, 0><<<grid(NT, 384, 1), 256, 0, stream>>>(
      (const h16*)r4, 96, 0, w_m1, 96, 0, r1, 384, 0, mlp1_b, nullptr, 0,
      nullptr, nullptr, 1.f, NT, 384, 96);
  gemm_wmma<0, 1, 0, 1, 1, 0><<<grid(NT, 96, 1), 256, 0, stream>>>(
      (const h16*)r1, 384, 0, w_m2, 384, 0, r3, 96, 0, mlp2_b, (const float*)r2, 96,
      nullptr, nullptr, 1.f, NT, 96, 384);

  // ---------------- global branch ----------------
  k_pool<<<(96 * 1728) / 256, 256, 0, stream>>>(x, tg);
  k_ln_rows<<<GP / 8, 256, 0, stream>>>(tg, g_ln1_w, g_ln1_b, ag16, GP);
  gemm_wmma<1, 1, 0, 1, 0, 0><<<grid(GP, 96, 1), 256, 0, stream>>>(
      ag16, 96, 0, w_q, 96, 0, qg, 96, 0, q_b, nullptr, 0, nullptr, nullptr, 1.f, GP, 96, 96);
  gemm_wmma<1, 1, 0, 1, 0, 0><<<grid(GP, 96, 1), 256, 0, stream>>>(
      ag16, 96, 0, w_k, 96, 0, kg, 96, 0, k_b, nullptr, 0, nullptr, nullptr, 1.f, GP, 96, 96);
  gemm_wmma<1, 1, 0, 1, 0, 0><<<grid(GP, 96, 1), 256, 0, stream>>>(
      ag16, 96, 0, w_v, 96, 0, vg, 96, 0, v_b, nullptr, 0, nullptr, nullptr, 1.f, GP, 96, 96);
  gemm_wmma<0, 1, 0, 0, 0, 0><<<grid(GP, GP, 4), 256, 0, stream>>>(
      qg, 96, 24, kg, 96, 24, r2, GP, (long)GP * GP, nullptr, nullptr, 0,
      nullptr, nullptr, 0.20412414523193154f, GP, GP, 24);
  k_softmax<<<4 * GP, 256, 0, stream>>>((const float*)r2, (h16*)r4);
  gemm_wmma<1, 0, 0, 0, 0, 0><<<grid(GP, 24, 4), 256, 0, stream>>>(
      (const h16*)r4, GP, (long)GP * GP, vg, 96, 24, ago, 96, 24, nullptr,
      nullptr, 0, nullptr, nullptr, 1.f, GP, 24, GP);
  gemm_wmma<0, 1, 0, 1, 1, 0><<<grid(GP, 96, 1), 256, 0, stream>>>(
      ago, 96, 0, w_o, 96, 0, tg2, 96, 0, o_b, tg, 96, nullptr, nullptr, 1.f, GP, 96, 96);
  k_ln_rows<<<GP / 8, 256, 0, stream>>>(tg2, g_ln2_w, g_ln2_b, a2, GP);
  gemm_wmma<1, 1, 1, 1, 0, 0><<<grid(GP, 96, 1), 256, 0, stream>>>(
      a2, 96, 0, w_f1, 96, 0, h2, 96, 0, g_fc1_b, nullptr, 0, nullptr, nullptr, 1.f, GP, 96, 96);
  gemm_wmma<0, 1, 0, 1, 1, 0><<<grid(GP, 96, 1), 256, 0, stream>>>(
      h2, 96, 0, w_f2, 96, 0, tg3, 96, 0, g_fc2_b, tg2, 96, nullptr, nullptr, 1.f, GP, 96, 96);

  // ---------------- fuse ----------------
  k_cat<<<NT / 256, 256, 0, stream>>>((const float*)r3, tg3, (h16*)r1);
  gemm_wmma<0, 1, 0, 1, 0, 1><<<grid(NT, 96, 1), 256, 0, stream>>>(
      (const h16*)r1, 192, 0, w_cv, 192, 0, d_out, 0, 0, conv_b, nullptr, 0,
      bn_g, bn_b, 1.f, NT, 96, 192);
}
